// MultiHeadSelfAttention_26190710571235
// MI455X (gfx1250) — matrix operand order repaired and checked by reading, not compiled, not run on hardware
//
#include <hip/hip_runtime.h>

// ---------------------------------------------------------------------------
// MI455X (gfx1250) multi-head self-attention.
//   bf16 WMMA everywhere + TDM (tensor_load_to_lds) tile staging + flash attn.
//   B=4, S=2048, D=1024, H=16, Hd=64.
// ---------------------------------------------------------------------------

typedef __attribute__((ext_vector_type(16))) __bf16  v16bf;
typedef __attribute__((ext_vector_type(8)))  float   v8f;
typedef __attribute__((ext_vector_type(4)))  unsigned int u32x4;

#define WMMA_BF16(a, b, c) \
  __builtin_amdgcn_wmma_f32_16x16x32_bf16(false, (a), false, (b), (short)0, (c), false, false)

#if defined(__gfx1250__) && __has_builtin(__builtin_amdgcn_tensor_load_to_lds) && \
    __has_builtin(__builtin_amdgcn_s_wait_tensorcnt)
#define HAVE_TDM 1
#else
#define HAVE_TDM 0
#endif

#if HAVE_TDM
typedef __attribute__((ext_vector_type(4))) unsigned int tdm_g0_t;
typedef __attribute__((ext_vector_type(8))) int          tdm_g1_t;
typedef __attribute__((ext_vector_type(4))) int          tdm_g4_t;

typedef __attribute__((address_space(3))) void lds_void;

// LDS byte offset of a generic pointer into __shared__ (AS3 ptrtoint = offset).
__device__ inline unsigned lds_off_of(const void* p) {
  return (unsigned)(unsigned long long)(lds_void*)(p);
}

// TDM 2D tile load: tile_x elements per row (bf16), tile_rows rows,
// global row stride row_stride (elements).  Dense pack into LDS at lds_off.
// D# per CDNA5 ISA (group0 128b, group1 256b).
__device__ inline void tdm_load_2d(unsigned lds_off, const void* gptr,
                                   unsigned tile_x, unsigned tile_rows,
                                   unsigned row_stride) {
  const unsigned long long ga = (unsigned long long)(size_t)gptr;
  tdm_g0_t g0;
  g0[0] = 1u;                                   // count=1, user mode
  g0[1] = lds_off;                              // lds_addr (bytes)
  g0[2] = (unsigned)(ga & 0xffffffffu);         // global_addr[31:0]
  g0[3] = (unsigned)((ga >> 32) & 0x1ffffffu)   // global_addr[56:32]
          | (2u << 30);                         // type=2 ("image")
  const unsigned td0 = row_stride;              // tensor_dim0 >= tile_x
  const unsigned td1 = tile_rows;               // tensor_dim1 >= tile_rows
  tdm_g1_t g1;
  g1[0] = (int)(1u << 16);                      // data_size=1 (2 bytes)
  g1[1] = (int)((td0 & 0xffffu) << 16);         // tensor_dim0[15:0]
  g1[2] = (int)((td0 >> 16) | ((td1 & 0xffffu) << 16));   // td0[31:16], td1[15:0]
  g1[3] = (int)((td1 >> 16) | (tile_x << 16));  // td1[31:16], tile_dim0
  g1[4] = (int)tile_rows;                       // tile_dim1 (tile_dim2=0)
  g1[5] = (int)row_stride;                      // tensor_dim0_stride[31:0]
  g1[6] = 0;                                    // stride hi / dim1_stride lo
  g1[7] = 0;
  tdm_g4_t gz4 = {0, 0, 0, 0};                  // <=2D: groups 2/3 unused
  tdm_g1_t gz8 = {0, 0, 0, 0, 0, 0, 0, 0};
  __builtin_amdgcn_tensor_load_to_lds(g0, g1, gz4, gz4, gz8, 0);
}
#endif  // HAVE_TDM

// ---- WMMA fragment loaders (wave32; contiguous k = 16*(lane/16) + s, same order in A and B) ----

__device__ inline v16bf frag_a_bf16(const __bf16* A, int lda) {
  const int lane = threadIdx.x & 31;
  const int m = lane & 15, g = lane >> 4;
  v16bf a;
#pragma unroll
  for (int j = 0; j < 8; ++j) {
    const int kk = 16 * g + 2 * j;
    a[2 * j]     = A[m * lda + kk];
    a[2 * j + 1] = A[m * lda + kk + 1];
  }
  return a;
}

__device__ inline v16bf frag_a_f32(const float* A, int lda) {
  const int lane = threadIdx.x & 31;
  const int m = lane & 15, g = lane >> 4;
  v16bf a;
#pragma unroll
  for (int j = 0; j < 8; ++j) {
    const int kk = 16 * g + 2 * j;
    a[2 * j]     = (__bf16)A[m * lda + kk];
    a[2 * j + 1] = (__bf16)A[m * lda + kk + 1];
  }
  return a;
}

__device__ inline v16bf frag_b_nk(const __bf16* Bt, int ldb) {
  const int lane = threadIdx.x & 31;
  const int n = lane & 15, g = lane >> 4;
  v16bf b;
#pragma unroll
  for (int j = 0; j < 8; ++j) {
    const int k = 16 * g + 2 * j;
    b[2 * j]     = Bt[n * ldb + k];
    b[2 * j + 1] = Bt[n * ldb + k + 1];
  }
  return b;
}

__device__ inline v16bf frag_b_kn(const __bf16* Bm, int ldb) {
  const int lane = threadIdx.x & 31;
  const int n = lane & 15, g = lane >> 4;
  v16bf b;
#pragma unroll
  for (int j = 0; j < 8; ++j) {
    const int k = 16 * g + 2 * j;
    b[2 * j]     = Bm[k * ldb + n];
    b[2 * j + 1] = Bm[(k + 1) * ldb + n];
  }
  return b;
}

// ---------------------------------------------------------------------------
// Kernel 0: fp32 -> bf16 conversion
// ---------------------------------------------------------------------------
__global__ __launch_bounds__(256) void cvt_f32_bf16(const float* __restrict__ src,
                                                    __bf16* __restrict__ dst, int n) {
  int i = blockIdx.x * 256 + threadIdx.x;
  int stride = gridDim.x * 256;
  for (; i < n; i += stride) dst[i] = (__bf16)src[i];
}

// ---------------------------------------------------------------------------
// Shared GEMM body: C[128x128] macro tile, 8 waves (4 M x 2 N),
// wave tile 32x64 = 2x4 WMMA tiles, K-loop step 32, TDM double buffering.
// ---------------------------------------------------------------------------
#define GEMM_PROLOGUE_AND_LOOP(A_PTR, B_PTR, LDK)                               \
  __shared__ __attribute__((aligned(16))) __bf16 sA[2][128 * 32];               \
  __shared__ __attribute__((aligned(16))) __bf16 sB[2][128 * 32];               \
  const int t = threadIdx.x;                                                    \
  const int wave = t >> 5, lane = t & 31;                                       \
  const int wm = wave & 3, wn = wave >> 2;                                      \
  const int g = lane >> 4, nl = lane & 15;                                      \
  (void)g; (void)nl;                                                            \
  v8f acc[2][4] = {};                                                           \
  TDM_PROLOGUE(A_PTR, B_PTR, LDK)                                               \
  for (int kt = 0, buf = 0; kt < (LDK); kt += 32, buf ^= 1) {                   \
    STAGE_STEP(A_PTR, B_PTR, LDK)                                               \
    __syncthreads();                                                            \
    const __bf16* curA = sA[CUR_BUF];                                           \
    const __bf16* curB = sB[CUR_BUF];                                           \
    v16bf bf[4];                                                                \
    _Pragma("unroll")                                                           \
    for (int j = 0; j < 4; ++j) bf[j] = frag_b_nk(curB + (64 * wn + 16 * j) * 32, 32); \
    _Pragma("unroll")                                                           \
    for (int i = 0; i < 2; ++i) {                                               \
      v16bf a = frag_a_bf16(curA + (32 * wm + 16 * i) * 32, 32);                \
      _Pragma("unroll")                                                         \
      for (int j = 0; j < 4; ++j) acc[i][j] = WMMA_BF16(a, bf[j], acc[i][j]);   \
    }                                                                           \
    __syncthreads();                                                            \
  }

#if HAVE_TDM
#define CUR_BUF buf
#define TDM_PROLOGUE(A_PTR, B_PTR, LDK)                                         \
  if (t < 32) {                                                                 \
    tdm_load_2d(lds_off_of(sA[0]), (A_PTR) + (size_t)gm0 * (LDK), 32, 128, (LDK)); \
    tdm_load_2d(lds_off_of(sB[0]), (B_PTR) + (size_t)gn0 * (LDK), 32, 128, (LDK)); \
  }
#define STAGE_STEP(A_PTR, B_PTR, LDK)                                           \
  if (t < 32) {                                                                 \
    if (kt + 32 < (LDK)) {                                                      \
      tdm_load_2d(lds_off_of(sA[buf ^ 1]),                                      \
                  (A_PTR) + (size_t)gm0 * (LDK) + kt + 32, 32, 128, (LDK));     \
      tdm_load_2d(lds_off_of(sB[buf ^ 1]),                                      \
                  (B_PTR) + (size_t)gn0 * (LDK) + kt + 32, 32, 128, (LDK));     \
      __builtin_amdgcn_s_wait_tensorcnt(2); /* retire current pair */           \
    } else {                                                                    \
      __builtin_amdgcn_s_wait_tensorcnt(0);                                     \
    }                                                                           \
  }
#else
#define CUR_BUF 0
#define TDM_PROLOGUE(A_PTR, B_PTR, LDK)
#define STAGE_STEP(A_PTR, B_PTR, LDK)                                           \
  {                                                                             \
    const int row = t >> 1, c = (t & 1) * 16;                                   \
    const u32x4* s = (const u32x4*)((A_PTR) + (size_t)(gm0 + row) * (LDK) + kt + c); \
    u32x4* d = (u32x4*)(sA[0] + row * 32 + c);                                  \
    d[0] = s[0]; d[1] = s[1];                                                   \
    const u32x4* s2 = (const u32x4*)((B_PTR) + (size_t)(gn0 + row) * (LDK) + kt + c); \
    u32x4* d2 = (u32x4*)(sB[0] + row * 32 + c);                                 \
    d2[0] = s2[0]; d2[1] = s2[1];                                               \
  }
#endif

// ---------------------------------------------------------------------------
// Kernel 1: QKV projection (M=8192, N=3072, K=1024) -> scatter Q/K/V bf16
// ---------------------------------------------------------------------------
__global__ __launch_bounds__(256) void qkv_gemm(const __bf16* __restrict__ xb,
                                                const __bf16* __restrict__ wb,
                                                const float* __restrict__ bias,
                                                __bf16* __restrict__ Qb,
                                                __bf16* __restrict__ Kb,
                                                __bf16* __restrict__ Vb) {
  const int gn0 = blockIdx.x * 128;
  const int gm0 = blockIdx.y * 128;
  GEMM_PROLOGUE_AND_LOOP(xb, wb, 1024)

#pragma unroll
  for (int i = 0; i < 2; ++i)
#pragma unroll
    for (int j = 0; j < 4; ++j)
#pragma unroll
      for (int r = 0; r < 8; ++r) {
        const int mg = gm0 + 32 * wm + 16 * i + 8 * g + r;
        const int ng = gn0 + 64 * wn + 16 * j + nl;
        float v = acc[i][j][r] + bias[ng];
        const int which = ng >> 10;           // 0=Q 1=K 2=V
        const int h = (ng >> 6) & 15;
        const int d = ng & 63;
        const int b = mg >> 11, s = mg & 2047;
        if (which == 0) v *= 0.125f;          // fold 1/sqrt(64) into Q
        __bf16* dst = (which == 0) ? Qb : (which == 1) ? Kb : Vb;
        dst[(((size_t)(b * 16 + h)) * 2048 + s) * 64 + d] = (__bf16)v;
      }
}

// ---------------------------------------------------------------------------
// Kernel 2: flash attention (block = one (b*h, 64-query tile))
// ---------------------------------------------------------------------------
__global__ __launch_bounds__(256) void attn_flash(const __bf16* __restrict__ Qb,
                                                  const __bf16* __restrict__ Kb,
                                                  const __bf16* __restrict__ Vb,
                                                  __bf16* __restrict__ attb) {
  __shared__ __attribute__((aligned(16))) __bf16 sQ[64 * 64];
  __shared__ __attribute__((aligned(16))) __bf16 sK[64 * 64];
  __shared__ __attribute__((aligned(16))) __bf16 sV[64 * 64];
  __shared__ __attribute__((aligned(16))) float  sS[64 * 64];
  __shared__ __attribute__((aligned(16))) float  sO[64 * 64];
  __shared__ float sMax[64], sSum[64], sAlpha[64];

  const int t = threadIdx.x;
  const int bh = blockIdx.y;
  const int q0 = blockIdx.x * 64;
  const size_t base = (size_t)bh * 2048 * 64;
  const int wave = t >> 5, lane = t & 31;
  const int wm = wave & 3, wn = wave >> 2;
  const int g = lane >> 4, nl = lane & 15;

#if HAVE_TDM
  if (t < 32)  // Q tile: contiguous 4096 bf16
    tdm_load_2d(lds_off_of(sQ), Qb + base + (size_t)q0 * 64, 4096, 1, 4096);
#else
  {
    const int q = t >> 2, c = (t & 3) * 16;
    const u32x4* s = (const u32x4*)(Qb + base + (size_t)(q0 + q) * 64 + c);
    u32x4* d = (u32x4*)(sQ + q * 64 + c);
    d[0] = s[0]; d[1] = s[1];
  }
#endif
  {
    const int idx = t * 16;
#pragma unroll
    for (int i = 0; i < 16; ++i) sO[idx + i] = 0.0f;
  }
  if (t < 64) { sMax[t] = -1e30f; sSum[t] = 0.0f; }

  for (int j0 = 0; j0 < 2048; j0 += 64) {
#if HAVE_TDM
    if (t < 32) {
      tdm_load_2d(lds_off_of(sK), Kb + base + (size_t)j0 * 64, 4096, 1, 4096);
      tdm_load_2d(lds_off_of(sV), Vb + base + (size_t)j0 * 64, 4096, 1, 4096);
      __builtin_amdgcn_s_wait_tensorcnt(0);
    }
#else
    {
      const int r = t >> 2, c = (t & 3) * 16;
      const u32x4* sk = (const u32x4*)(Kb + base + (size_t)(j0 + r) * 64 + c);
      u32x4* dk = (u32x4*)(sK + r * 64 + c);
      dk[0] = sk[0]; dk[1] = sk[1];
      const u32x4* sv = (const u32x4*)(Vb + base + (size_t)(j0 + r) * 64 + c);
      u32x4* dv = (u32x4*)(sV + r * 64 + c);
      dv[0] = sv[0]; dv[1] = sv[1];
    }
#endif
    __syncthreads();

    // scores S = Q @ K^T : wave computes 16x32 patch; Hd=64 -> 2 K-steps
    v8f sc[2] = {};
#pragma unroll
    for (int d0 = 0; d0 < 64; d0 += 32) {
      v16bf a = frag_a_bf16(sQ + 16 * wm * 64 + d0, 64);
#pragma unroll
      for (int j = 0; j < 2; ++j) {
        v16bf b = frag_b_nk(sK + (32 * wn + 16 * j) * 64 + d0, 64);
        sc[j] = WMMA_BF16(a, b, sc[j]);
      }
    }
#pragma unroll
    for (int j = 0; j < 2; ++j)
#pragma unroll
      for (int r = 0; r < 8; ++r)
        sS[(16 * wm + 8 * g + r) * 64 + 32 * wn + 16 * j + nl] = sc[j][r];
    __syncthreads();

    // online softmax: one thread per query row
    if (t < 64) {
      float mold = sMax[t];
      float mnew = mold;
      for (int j = 0; j < 64; ++j) mnew = fmaxf(mnew, sS[t * 64 + j]);
      const float alpha = __expf(mold - mnew);
      float psum = 0.0f;
      for (int j = 0; j < 64; ++j) {
        const float p = __expf(sS[t * 64 + j] - mnew);
        sS[t * 64 + j] = p;
        psum += p;
      }
      sMax[t] = mnew;
      sSum[t] = sSum[t] * alpha + psum;
      sAlpha[t] = alpha;
    }
    __syncthreads();

    // O = O*alpha + P@V
    v8f pv[2] = {};
#pragma unroll
    for (int kk = 0; kk < 64; kk += 32) {
      v16bf a = frag_a_f32(sS + 16 * wm * 64 + kk, 64);
#pragma unroll
      for (int j = 0; j < 2; ++j) {
        v16bf b = frag_b_kn(sV + kk * 64 + 32 * wn + 16 * j, 64);
        pv[j] = WMMA_BF16(a, b, pv[j]);
      }
    }
#pragma unroll
    for (int j = 0; j < 2; ++j)
#pragma unroll
      for (int r = 0; r < 8; ++r) {
        const int row = 16 * wm + 8 * g + r;
        const int idx = row * 64 + 32 * wn + 16 * j + nl;
        sO[idx] = sO[idx] * sAlpha[row] + pv[j][r];
      }
    __syncthreads();
  }

  const int bb = bh >> 4, h = bh & 15;
  {
    const int q = t >> 2, c = (t & 3) * 16;
#pragma unroll
    for (int i = 0; i < 16; ++i) {
      const int d = c + i;
      attb[((size_t)(bb * 2048 + q0 + q)) * 1024 + h * 64 + d] =
          (__bf16)(sO[q * 64 + d] / sSum[q]);
    }
  }
}

// ---------------------------------------------------------------------------
// Kernel 3: output projection (M=8192, N=1024, K=1024) -> fp32 out + bias
// ---------------------------------------------------------------------------
__global__ __launch_bounds__(256) void oproj_gemm(const __bf16* __restrict__ ab,
                                                  const __bf16* __restrict__ wb,
                                                  const float* __restrict__ bias,
                                                  float* __restrict__ out) {
  const int gn0 = blockIdx.x * 128;
  const int gm0 = blockIdx.y * 128;
  GEMM_PROLOGUE_AND_LOOP(ab, wb, 1024)

#pragma unroll
  for (int i = 0; i < 2; ++i)
#pragma unroll
    for (int j = 0; j < 4; ++j)
#pragma unroll
      for (int r = 0; r < 8; ++r) {
        const int mg = gm0 + 32 * wm + 16 * i + 8 * g + r;
        const int ng = gn0 + 64 * wn + 16 * j + nl;
        out[(size_t)mg * 1024 + ng] = acc[i][j][r] + bias[ng];
      }
}

// ---------------------------------------------------------------------------
// Host-side launch.  Workspace (bytes):
//   [0)        x bf16 16MB | [16777216) qkv_w bf16 6MB | [23068672) out_w bf16 2MB
//   [25165824) Q bf16 16MB | [41943040) K bf16 16MB    | [58720256) V bf16 16MB
//   [75497472) attended bf16 16MB       total 92,274,688 bytes
// ---------------------------------------------------------------------------
extern "C" void kernel_launch(void* const* d_in, const int* in_sizes, int n_in,
                              void* d_out, int out_size, void* d_ws, size_t ws_size,
                              hipStream_t stream) {
  const float* x     = (const float*)d_in[0];
  const float* qkv_w = (const float*)d_in[1];
  const float* qkv_b = (const float*)d_in[2];
  const float* out_w = (const float*)d_in[3];
  const float* out_b = (const float*)d_in[4];
  float* out         = (float*)d_out;

  char* ws = (char*)d_ws;
  __bf16* xb   = (__bf16*)(ws + 0);
  __bf16* wqb  = (__bf16*)(ws + 16777216);
  __bf16* wob  = (__bf16*)(ws + 23068672);
  __bf16* Qb   = (__bf16*)(ws + 25165824);
  __bf16* Kb   = (__bf16*)(ws + 41943040);
  __bf16* Vb   = (__bf16*)(ws + 58720256);
  __bf16* attb = (__bf16*)(ws + 75497472);

  cvt_f32_bf16<<<dim3(8192), dim3(256), 0, stream>>>(x, xb, 8388608);
  cvt_f32_bf16<<<dim3(4096), dim3(256), 0, stream>>>(qkv_w, wqb, 3145728);
  cvt_f32_bf16<<<dim3(2048), dim3(256), 0, stream>>>(out_w, wob, 1048576);

  // QKV projection: grid = (N/128, M/128) = (24, 64)
  qkv_gemm<<<dim3(24, 64), dim3(256), 0, stream>>>(xb, wqb, qkv_b, Qb, Kb, Vb);

  // flash attention: grid = (S/64, B*H) = (32, 64)
  attn_flash<<<dim3(32, 64), dim3(256), 0, stream>>>(Qb, Kb, Vb, attb);

  // output projection: grid = (N/128, M/128) = (8, 64)
  oproj_gemm<<<dim3(8, 64), dim3(256), 0, stream>>>(attb, wob, out_b, out);
}
